// RoutingFreeGate_34643206210297
// MI455X (gfx1250) — compile-verified, matrix operand order checked by reading, not run on hardware
//
#include <hip/hip_runtime.h>
#include <hip/hip_bf16.h>
#include <math.h>

// ---------------------------------------------------------------------------
// RoutingFreeGate for MI455X (gfx1250, wave32, WMMA + Tensor Data Mover).
//
// C[token, r] = sum_k x[token,k] * W_A[r,k]   (tall-skinny GEMM, R=64)
// score = sqrt(sum_r C^2) * scale + bias ; gate = mask && (score >= 0.5)
//
// Memory-bound: 256 MB of x @ 23.3 TB/s ~= 11.5 us floor. x tiles are DMA'd
// into LDS by the TDM (tensor_load_to_lds, TENSORcnt) with HW row padding so
// ds_load_b128 fragment reads are bank-conflict-free; fp32->bf16 conversion
// happens at fragment read; matrix math via v_wmma_f32_16x16x32_bf16.
// ---------------------------------------------------------------------------

typedef __attribute__((ext_vector_type(16))) __bf16       v16bf;
typedef __attribute__((ext_vector_type(8)))  __bf16       v8bf;
typedef __attribute__((ext_vector_type(8)))  float        v8f;
typedef __attribute__((ext_vector_type(4)))  unsigned int v4u;
typedef __attribute__((ext_vector_type(8)))  int          v8i;
typedef __attribute__((ext_vector_type(4)))  int          v4i;

#define HIDDEN    2048
#define RANK      64
#define M_TILE    128            // tokens per workgroup (8 waves x 16 rows)
#define K_TILE    32             // x K-step per barrier (one WMMA K-step)
#define N_TILES   (HIDDEN / K_TILE)      // 64 K-steps
#define K_CHUNK   128            // W_A K-chunk resident in LDS (4 tiles)
#define W_STRIDE  136            // 128+8 bf16: 68 dwords/row -> conflict-free
#define A_STRIDE  36             // 32+4 fp32 dwords/row -> conflict-free (TDM pad)

#define W_BYTES   (RANK * W_STRIDE * 2)          // 17,408
#define A_BYTES   (M_TILE * A_STRIDE * 4)        // 18,432
#define A0_OFF    W_BYTES
#define A1_OFF    (W_BYTES + A_BYTES)
#define SMEM_BYTES (W_BYTES + 2 * A_BYTES)       // 54,272 B

#if __has_builtin(__builtin_amdgcn_tensor_load_to_lds) && \
    __has_builtin(__builtin_amdgcn_s_wait_tensorcnt)
#define USE_TDM 1
#else
#define USE_TDM 0
#endif

#if USE_TDM
// Issue a TDM load of x tile [row0..row0+127] x [t*32..t*32+31] (fp32) into
// LDS at byte offset lds_off, padded 4 dwords per 32-dword row (stride 36).
__device__ __forceinline__ void tdm_issue_tile(const float* x, int row0, int t,
                                               unsigned lds_off) {
    unsigned long long ga =
        (unsigned long long)(const void*)(x + (size_t)row0 * HIDDEN + t * K_TILE);
    // D# group 0: count=1 | lds_addr | global_addr[56:0] | type=2
    v4u g0 = {1u,
              lds_off,
              (unsigned)ga,
              (unsigned)((ga >> 32) & 0x1FFFFFFu) | (2u << 30)};
    // D# group 1: data_size=4B(2), pad_enable, pad_interval=32dw(4),
    // pad_amount=4dw(3); tensor_dim0=2048, tensor_dim1=128,
    // tile_dim0=32, tile_dim1=128, tensor_dim0_stride=2048.
    v8i g1 = {(2 << 16) | (1 << 20) | (4 << 22) | (3 << 25),
              (HIDDEN & 0xFFFF) << 16,
              (HIDDEN >> 16) | ((M_TILE & 0xFFFF) << 16),
              (K_TILE << 16),
              M_TILE,
              HIDDEN,
              0, 0};
    v4i z4 = {0, 0, 0, 0};
#if __clang_major__ >= 23
    v8i z8 = {0, 0, 0, 0, 0, 0, 0, 0};
    __builtin_amdgcn_tensor_load_to_lds(g0, g1, z4, z4, z8, 0);
#else
    __builtin_amdgcn_tensor_load_to_lds(g0, g1, z4, z4, 0);
#endif
}
#endif

__global__ __launch_bounds__(256)
void RoutingFreeGate_kernel(const float* __restrict__ x,
                            const unsigned char* __restrict__ mask,
                            const float* __restrict__ W_A,
                            const float* __restrict__ gate_scale,
                            const float* __restrict__ gate_bias,
                            float* __restrict__ out,
                            int n_tokens)
{
    // One flat shared block: TDM descriptor lds_addr uses static byte offsets.
    __shared__ __align__(128) unsigned char smem[SMEM_BYTES];
    __bf16* Wl = (__bf16*)smem;
    // NOTE: no pointer arrays holding LDS addresses (addrspacecast in a
    // constant initializer breaks ld.lld) — select buffers via ternary.
    float* A0 = (float*)(smem + A0_OFF);
    float* A1 = (float*)(smem + A1_OFF);

    const int tid  = threadIdx.x;
    const int lane = tid & 31;           // wave32
    const int wave = tid >> 5;           // 0..7
    const int ln15 = lane & 15;
    const int hi   = lane >> 4;          // selects K sub-slice per ISA A/B layout
    const int row0 = blockIdx.x * M_TILE;

    v8f acc[4] = {v8f{}, v8f{}, v8f{}, v8f{}};   // C: 16 rows x 64 cols per wave

#if USE_TDM
    if (wave == 0) tdm_issue_tile(x, row0, 0, (unsigned)A0_OFF);   // prologue DMA
#endif

    for (int t = 0; t < N_TILES; ++t) {
        // ---- stage W_A chunk (RANK x K_CHUNK) fp32 -> bf16 every 4 tiles ----
        if ((t & 3) == 0) {
            __syncthreads();             // previous chunk's compute done
            const int kc   = (t >> 2) * K_CHUNK;
            const int c4   = tid & 31;   // float4 column (32 per 128-wide row)
            const int rb   = tid >> 5;   // 0..7
            #pragma unroll
            for (int p = 0; p < 8; ++p) {
                const int r = p * 8 + rb;               // 0..63
                const float4 wv =
                    *(const float4*)(W_A + (size_t)r * HIDDEN + kc + c4 * 4);
                __bf16* dst = Wl + r * W_STRIDE + c4 * 4;
                dst[0] = (__bf16)wv.x; dst[1] = (__bf16)wv.y;
                dst[2] = (__bf16)wv.z; dst[3] = (__bf16)wv.w;
            }
        }

        float* Acur = (t & 1) ? A1 : A0;

#if USE_TDM
        if (wave == 0) __builtin_amdgcn_s_wait_tensorcnt(0);  // tile t landed
        __syncthreads();                 // all waves see tile t + W chunk
        if (wave == 0 && t + 1 < N_TILES) {                   // DMA next tile,
            const unsigned next_off = ((t + 1) & 1) ? (unsigned)A1_OFF
                                                    : (unsigned)A0_OFF;
            tdm_issue_tile(x, row0, t + 1, next_off);         // overlaps compute
        }
#else
        __syncthreads();                 // prev compute done before overwrite
        {   // fallback: cooperative fp32 tile load (same LDS layout)
            const int c4 = tid & 7;      // 8 float4 per 32-wide row
            const int rb = tid >> 3;     // 0..31
            #pragma unroll
            for (int p = 0; p < 4; ++p) {
                const int r = p * 32 + rb;
                const float* src = x + (size_t)(row0 + r) * HIDDEN + t * K_TILE + c4 * 4;
                const float4 av = *(const float4*)src;
                if (t + 1 < N_TILES)
                    __builtin_prefetch(src + K_TILE, 0, 0);    // global_prefetch_b8
                *(float4*)(Acur + r * A_STRIDE + c4 * 4) = av;
            }
        }
        __syncthreads();
#endif

        // ---- compute one 32-wide K-step: 4 N-subtiles ----
        // ISA 16-bit A layout: lanes 0-15 hold K {0..7,16..23}, lanes 16-31
        // hold K {8..15,24..31}; fp32 read from LDS, cvt_pk to bf16 here.
        const int koff = hi * 8;
        v16bf a;
        {
            const float* ap = Acur + (wave * 16 + ln15) * A_STRIDE + koff;
            const float4 f0 = *(const float4*)(ap + 0);    // ds_load_b128 x4
            const float4 f1 = *(const float4*)(ap + 4);
            const float4 f2 = *(const float4*)(ap + 16);
            const float4 f3 = *(const float4*)(ap + 20);
            a[0]=(__bf16)f0.x;  a[1]=(__bf16)f0.y;  a[2]=(__bf16)f0.z;  a[3]=(__bf16)f0.w;
            a[4]=(__bf16)f1.x;  a[5]=(__bf16)f1.y;  a[6]=(__bf16)f1.z;  a[7]=(__bf16)f1.w;
            a[8]=(__bf16)f2.x;  a[9]=(__bf16)f2.y;  a[10]=(__bf16)f2.z; a[11]=(__bf16)f2.w;
            a[12]=(__bf16)f3.x; a[13]=(__bf16)f3.y; a[14]=(__bf16)f3.z; a[15]=(__bf16)f3.w;
        }
        const int wk = (t & 3) * K_TILE + koff;   // chunk-local W column
        #pragma unroll
        for (int nt = 0; nt < 4; ++nt) {
            const __bf16* bp = Wl + (nt * 16 + ln15) * W_STRIDE + wk;
            const v8bf blo = *(const v8bf*)bp;             // ds_load_b128 x2
            const v8bf bhi = *(const v8bf*)(bp + 16);
            v16bf b;
            #pragma unroll
            for (int i = 0; i < 8; ++i) { b[i] = blo[i]; b[8 + i] = bhi[i]; }
            acc[nt] = __builtin_amdgcn_wmma_f32_16x16x32_bf16(
                /*neg_a=*/false, a, /*neg_b=*/false, b,
                /*c_mod=*/(short)0, acc[nt],
                /*reuse_a=*/false, /*reuse_b=*/false);
        }
    }

    // ---- epilogue: per-row sum of squares over 64 rank columns ----
    // C/D layout: VGPR v holds row M=v (lanes 0-15) / M=v+8 (lanes 16-31),
    // column N = lane%16 within each 16-wide N-subtile.
    float s[8];
    #pragma unroll
    for (int v = 0; v < 8; ++v) {
        float tq = acc[0][v] * acc[0][v] + acc[1][v] * acc[1][v] +
                   acc[2][v] * acc[2][v] + acc[3][v] * acc[3][v];
        tq += __shfl_xor(tq, 1, 32);   // butterfly across the 16-lane half
        tq += __shfl_xor(tq, 2, 32);
        tq += __shfl_xor(tq, 4, 32);
        tq += __shfl_xor(tq, 8, 32);
        s[v] = tq;
    }

    if (ln15 == 0) {    // lanes 0 and 16 each publish 8 rows
        const float scale = gate_scale[0];
        const float bias  = gate_bias[0];
        #pragma unroll
        for (int v = 0; v < 8; ++v) {
            const int token = row0 + wave * 16 + hi * 8 + v;
            const float score = sqrtf(s[v]) * scale + bias;
            const bool g = (mask[token] != 0) && (score >= 0.5f);
            out[token]            = g ? 1.0f : 0.0f;                  // gate_mask_full
            out[n_tokens + token] = g ? score : -__builtin_inff();    // gate_score_full
        }
    }
}

extern "C" void kernel_launch(void* const* d_in, const int* in_sizes, int n_in,
                              void* d_out, int out_size, void* d_ws, size_t ws_size,
                              hipStream_t stream) {
    const float*         x    = (const float*)d_in[0];
    const unsigned char* mask = (const unsigned char*)d_in[1];   // jnp bool_ = 1 byte
    const float*         W_A  = (const float*)d_in[2];
    const float*         gs   = (const float*)d_in[3];
    const float*         gb   = (const float*)d_in[4];
    float*               out  = (float*)d_out;

    const int n_tokens = in_sizes[1];                 // 4 * 8192 = 32768
    const int grid     = n_tokens / M_TILE;           // 256 workgroups

    RoutingFreeGate_kernel<<<grid, 256, 0, stream>>>(x, mask, W_A, gs, gb, out, n_tokens);
}
